// Model_89773406421161
// MI455X (gfx1250) — matrix operand order repaired and checked by reading, not compiled, not run on hardware
//
#include <hip/hip_runtime.h>
#include <hip/hip_bf16.h>

typedef __attribute__((ext_vector_type(16))) _Float16 v16h;
typedef __attribute__((ext_vector_type(4)))  _Float16 v4h;
typedef __attribute__((ext_vector_type(8)))  float    v8f;

#define HDIM 64

// ---------------------------------------------------------------------------
// Pack a K x 64 fp32 weight matrix into f16 WMMA B-fragments.
// Fragment order: [kc][nt][lane][16 halves], 512 halves per (kc,nt) fragment.
// B layout (16-bit 32x16 tile, wave32): h=lane>>4, N=lane&15; idx<8 -> K=8h+idx,
// idx>=8 -> K=16+8h+(idx-8).  (CDNA5 ISA sect 7.12.5 analog)
// ---------------------------------------------------------------------------
__global__ void pack_b_kernel(const float* __restrict__ W,
                              _Float16* __restrict__ out, int total) {
    int t = blockIdx.x * blockDim.x + threadIdx.x;
    if (t >= total) return;
    int idx  = t & 15;
    int lane = (t >> 4) & 31;
    int nt   = (t >> 9) & 3;
    int kc   = t >> 11;
    int k = kc * 32 + ((lane >> 4) << 3) + (idx & 7) + ((idx >> 3) << 4);
    int n = nt * 16 + (lane & 15);
    out[t] = (_Float16)W[(size_t)k * HDIM + n];
}

// ---------------------------------------------------------------------------
// Tiled GEMM: out(16-row tile x 64) = A(tile x K) * Bpacked(K x 64) [+ bias]
// One wave per 16-row tile. A is fp32 row-major (lda), converted to f16 in
// registers per the 16-bit A 16x32 fragment layout. K = 32*kchunks.
// Exactly one of outF (fp32) / outH (f16) is non-null.
// ---------------------------------------------------------------------------
__global__ void wmma_gemm_kernel(const float* __restrict__ A, int lda, int kchunks,
                                 const _Float16* __restrict__ Bp,
                                 const float* __restrict__ bias,
                                 float* __restrict__ outF,
                                 _Float16* __restrict__ outH,
                                 int ntiles, int nrows) {
    int wave = blockIdx.x * (blockDim.x >> 5) + (threadIdx.x >> 5);
    if (wave >= ntiles) return;            // uniform per wave: EXEC stays all-1s
    int lane = threadIdx.x & 31;
    int hi   = lane >> 4;                  // half-wave select
    int m    = lane & 15;
    int row  = wave * 16 + m;
    int rowc = row < nrows ? row : (nrows - 1);   // clamp (N divisible by 16 here)
    const float* arow = A + (size_t)rowc * lda + hi * 8;

    v8f acc[4] = {};
#pragma unroll
    for (int kc = 0; kc < 4; ++kc) {
        if (kc >= kchunks) break;
        const float* p = arow + kc * 32;
        float4 f0 = *(const float4*)(p);
        float4 f1 = *(const float4*)(p + 4);
        float4 f2 = *(const float4*)(p + 16);
        float4 f3 = *(const float4*)(p + 20);
        v16h a;
        a[0]=(_Float16)f0.x; a[1]=(_Float16)f0.y; a[2]=(_Float16)f0.z; a[3]=(_Float16)f0.w;
        a[4]=(_Float16)f1.x; a[5]=(_Float16)f1.y; a[6]=(_Float16)f1.z; a[7]=(_Float16)f1.w;
        a[8]=(_Float16)f2.x; a[9]=(_Float16)f2.y; a[10]=(_Float16)f2.z; a[11]=(_Float16)f2.w;
        a[12]=(_Float16)f3.x; a[13]=(_Float16)f3.y; a[14]=(_Float16)f3.z; a[15]=(_Float16)f3.w;

        const _Float16* bbase = Bp + (size_t)kc * 2048 + lane * 16;
#pragma unroll
        for (int nt = 0; nt < 4; ++nt) {
            v16h b = *(const v16h*)(bbase + nt * 512);
            acc[nt] = __builtin_amdgcn_wmma_f32_16x16x32_f16(
                false, a, false, b, (short)0, acc[nt], false, false);
        }
    }

    // C/D layout: VGPR v -> row = tile*16 + v + 8*hi ; col = nt*16 + (lane&15)
    int mbase = wave * 16 + hi * 8;
#pragma unroll
    for (int nt = 0; nt < 4; ++nt) {
        int col = nt * 16 + m;
        float bv = bias ? bias[col] : 0.0f;
#pragma unroll
        for (int v = 0; v < 8; ++v) {
            int r = mbase + v;
            if (r >= nrows) continue;
            float val = acc[nt][v] + bv;
            if (outF) outF[(size_t)r * HDIM + col] = val;
            else      outH[(size_t)r * HDIM + col] = (_Float16)val;
        }
    }
}

// ---------------------------------------------------------------------------
// Edge scatter: agg[dst] += (float)xw[etype][src]  — 16 lanes per edge,
// 4 halves per lane, f32 atomics (agg is L2-resident: 25.6 MB << 192 MB L2).
// ---------------------------------------------------------------------------
__global__ void edge_scatter_kernel(const int* __restrict__ src,
                                    const int* __restrict__ dst,
                                    const int* __restrict__ et,
                                    const _Float16* __restrict__ xw,
                                    float* __restrict__ agg,
                                    int nE, int nNodes) {
    int e  = blockIdx.x * 16 + (threadIdx.x >> 4);
    int c4 = (threadIdx.x & 15) * 4;
    if (e >= nE) return;
    int s = src[e], d = dst[e], r = et[e];
    const v4h msg = *(const v4h*)(xw + ((size_t)r * nNodes + s) * HDIM + c4);
    float* ap = agg + (size_t)d * HDIM + c4;
    atomicAdd(ap + 0, (float)msg[0]);
    atomicAdd(ap + 1, (float)msg[1]);
    atomicAdd(ap + 2, (float)msg[2]);
    atomicAdd(ap + 3, (float)msg[3]);
}

__global__ void relu_kernel(const float* __restrict__ in, float* __restrict__ out, int n4) {
    int t = blockIdx.x * blockDim.x + threadIdx.x;
    if (t >= n4) return;
    float4 v = ((const float4*)in)[t];
    v.x = fmaxf(v.x, 0.0f); v.y = fmaxf(v.y, 0.0f);
    v.z = fmaxf(v.z, 0.0f); v.w = fmaxf(v.w, 0.0f);
    ((float4*)out)[t] = v;
}

__global__ void zero64_kernel(float* p) { p[threadIdx.x] = 0.0f; }

__global__ void pool_reduce_kernel(const float* __restrict__ x,
                                   float* __restrict__ pooled, int nRows) {
    __shared__ float sh[256];
    int col  = threadIdx.x & 63;
    int slot = threadIdx.x >> 6;
    float s = 0.0f;
    for (int r = blockIdx.x * 4 + slot; r < nRows; r += gridDim.x * 4)
        s += x[(size_t)r * HDIM + col];
    sh[threadIdx.x] = s;
    __syncthreads();
    if (slot == 0) {
        float t = sh[col] + sh[64 + col] + sh[128 + col] + sh[192 + col];
        atomicAdd(&pooled[col], t);
    }
}

__global__ void final_kernel(const float* __restrict__ pooled,
                             const float* __restrict__ Wd,
                             const float* __restrict__ bd,
                             float* __restrict__ out) {
    __shared__ float sh[64];
    int t = threadIdx.x;
    sh[t] = pooled[t] * Wd[t];
    __syncthreads();
    for (int s = 32; s >= 1; s >>= 1) {
        if (t < s) sh[t] += sh[t + s];
        __syncthreads();
    }
    if (t == 0) out[0] = sh[0] + bd[0];
}

// ---------------------------------------------------------------------------
extern "C" void kernel_launch(void* const* d_in, const int* in_sizes, int n_in,
                              void* d_out, int out_size, void* d_ws, size_t ws_size,
                              hipStream_t stream) {
    const float* h       = (const float*)d_in[0];
    const int*   src     = (const int*)  d_in[1];
    const int*   dst     = (const int*)  d_in[2];
    const int*   etypes  = (const int*)  d_in[3];
    const float* W_fc    = (const float*)d_in[4];
    const float* W_rels  = (const float*)d_in[5];   // (3,4,64,64)
    const float* W_loops = (const float*)d_in[6];   // (3,64,64)
    const float* biases  = (const float*)d_in[7];   // (3,64)
    const float* W_dense = (const float*)d_in[8];   // (64,1)
    const float* b_dense = (const float*)d_in[9];   // (1,)
    float* out = (float*)d_out;

    const int F_IN   = 128;
    const int nNodes = in_sizes[0] / F_IN;
    const int nEdges = in_sizes[1];
    const int nLayers = 3, nRels = 4;

    // Workspace layout (bytes)
    const size_t packedHalves = 8192 + (size_t)nLayers * 5 * 4096;   // fc + 15 mats
    size_t off_packed = 0;
    size_t off_x      = (packedHalves * 2 + 255) & ~(size_t)255;
    size_t xBytes     = (size_t)nNodes * HDIM * 4;
    size_t off_agg    = off_x + xBytes;
    size_t off_xw     = off_agg + xBytes;
    size_t xwBytes    = (size_t)nRels * nNodes * HDIM * 2;
    size_t off_pooled = off_xw + xwBytes;
    size_t need       = off_pooled + 256;
    if (ws_size < need) return;  // deterministic: ws_size fixed per harness

    _Float16* packed = (_Float16*)((char*)d_ws + off_packed);
    float*    x      = (float*)   ((char*)d_ws + off_x);
    float*    agg    = (float*)   ((char*)d_ws + off_agg);
    _Float16* xw     = (_Float16*)((char*)d_ws + off_xw);
    float*    pooled = (float*)   ((char*)d_ws + off_pooled);

    auto packOff = [&](int l, int w) -> size_t { return 8192 + (size_t)(l * 5 + w) * 4096; };

    // 1) Pack all weights into WMMA B-fragment layout (f16)
    pack_b_kernel<<<(8192 + 255) / 256, 256, 0, stream>>>(W_fc, packed, 8192);
    for (int l = 0; l < nLayers; ++l) {
        for (int r = 0; r < nRels; ++r)
            pack_b_kernel<<<16, 256, 0, stream>>>(W_rels + (size_t)(l * 4 + r) * 4096,
                                                  packed + packOff(l, r), 4096);
        pack_b_kernel<<<16, 256, 0, stream>>>(W_loops + (size_t)l * 4096,
                                              packed + packOff(l, 4), 4096);
    }

    const int ntiles = (nNodes + 15) / 16;
    const int gemmBlocks = (ntiles + 7) / 8;          // 8 waves per 256-thread block

    // 2) x = h @ W_fc   (K=128 -> 4 k-chunks)
    wmma_gemm_kernel<<<gemmBlocks, 256, 0, stream>>>(h, F_IN, 4, packed,
                                                     nullptr, x, nullptr, ntiles, nNodes);

    const int edgeBlocks = (nEdges + 15) / 16;
    const int n4 = nNodes * HDIM / 4;

    // 3) RelGraphConv stack
    for (int l = 0; l < nLayers; ++l) {
        // xw[r] = x @ W_rels[l][r]  (f16 messages, halves gather traffic)
        for (int r = 0; r < nRels; ++r)
            wmma_gemm_kernel<<<gemmBlocks, 256, 0, stream>>>(
                x, HDIM, 2, packed + packOff(l, r), nullptr,
                nullptr, xw + (size_t)r * nNodes * HDIM, ntiles, nNodes);
        // agg = x @ W_loops[l] + bias[l]   (fp32, overwrites agg)
        wmma_gemm_kernel<<<gemmBlocks, 256, 0, stream>>>(
            x, HDIM, 2, packed + packOff(l, 4), biases + (size_t)l * HDIM,
            agg, nullptr, ntiles, nNodes);
        // agg[dst] += xw[etype][src]  (f32 atomics into L2-resident agg)
        edge_scatter_kernel<<<edgeBlocks, 256, 0, stream>>>(src, dst, etypes, xw,
                                                            agg, nEdges, nNodes);
        // x = relu(agg)
        relu_kernel<<<(n4 + 255) / 256, 256, 0, stream>>>(agg, x, n4);
    }

    // 4) Sum-pool over nodes, then dot with W_dense
    zero64_kernel<<<1, 64, 0, stream>>>(pooled);
    pool_reduce_kernel<<<512, 256, 0, stream>>>(x, pooled, nNodes);
    final_kernel<<<1, 64, 0, stream>>>(pooled, W_dense, b_dense, out);
    (void)out_size; (void)n_in;
}
